// NBiMamba2_81552839016899
// MI455X (gfx1250) — matrix operand order repaired and checked by reading, not compiled, not run on hardware
//
#include <hip/hip_runtime.h>
#include <hip/hip_bf16.h>
#include <math.h>

// ---------------- problem constants ----------------
#define D_MODEL   512
#define D_IN      1024
#define D_STATE   128
#define D_CONVK   4
#define D_INNER   2048
#define HEADDIM   64
#define NHEADS    32
#define CONV_DIM  2304            // D_INNER + 2*D_STATE
#define D_PROJ    4384            // 2*D_INNER + 2*D_STATE + NHEADS
#define BATCH     2
#define SEQ       4096
#define QCHUNK    128
#define NCHUNK    (SEQ / QCHUNK)  // 32
#define NTOK      (BATCH * SEQ)   // 8192

typedef __attribute__((ext_vector_type(16))) __bf16 v16bf;
typedef __attribute__((ext_vector_type(8)))  float  v8f;
typedef __attribute__((ext_vector_type(4)))  unsigned int v4u;
typedef __attribute__((ext_vector_type(8)))  int v8i_t;
typedef __attribute__((ext_vector_type(4)))  int v4i_t;

union BF16x16 { v16bf v; unsigned short u[16]; uint4 q[2]; };
union PK8 { uint4 q; unsigned short us[8]; };
union F4x2 { float f[8]; float4 v[2]; };

__device__ __forceinline__ unsigned short f2bf(float f) {
  unsigned int u = __float_as_uint(f);
  u += 0x7fffu + ((u >> 16) & 1u);        // round-to-nearest-even
  return (unsigned short)(u >> 16);
}
__device__ __forceinline__ float bf2f(unsigned short h) {
  return __uint_as_float(((unsigned int)h) << 16);
}
__device__ __forceinline__ v8f wmma_bf16(v16bf a, v16bf b, v8f c) {
  return __builtin_amdgcn_wmma_f32_16x16x32_bf16(false, a, false, b, (short)0, c,
                                                 false, false);
}

// ---- vectorized WMMA fragment loaders (2x ds_load_b128 each) ----
__device__ __forceinline__ v16bf frag_A(const unsigned short* s, int r0, int c0, int ld) {
  int lane = threadIdx.x & 31, half = lane >> 4, m = lane & 15;
  BF16x16 f;
  const unsigned short* base = s + (r0 + m) * ld + c0 + half * 8;
  f.q[0] = *(const uint4*)base;
  f.q[1] = *(const uint4*)(base + 16);
  return f.v;
}
__device__ __forceinline__ v16bf frag_Bc(const unsigned short* s, int k0, int n0, int ld) {
  int lane = threadIdx.x & 31, half = lane >> 4, n = lane & 15;
  BF16x16 f;
  const unsigned short* base = s + (n0 + n) * ld + k0 + half * 8;
  f.q[0] = *(const uint4*)base;
  f.q[1] = *(const uint4*)(base + 16);
  return f.v;
}

// ---- Tensor Data Mover: 2D tile (tile_rows x tile_w bf16) global -> LDS.
// pad_int_code/pad_amt_code add LDS row padding (D# pseudocode units):
//   pad interval = 2<<pad_int_code dwords, pad amount = pad_amt_code+1 dwords.
// tensor_dim1 = rows_avail -> hardware OOB zero-fill beyond it.
__device__ __forceinline__ void tdm_load_2d(unsigned int lds_off,
                                            unsigned long long gaddr,
                                            unsigned int tile_w,
                                            unsigned int tile_rows,
                                            unsigned int rows_avail,
                                            unsigned int stride_elems,
                                            unsigned int pad_int_code,
                                            unsigned int pad_amt_code) {
  v4u g0;
  g0[0] = 1u;                                          // count=1, user mode
  g0[1] = lds_off;                                     // lds_addr [63:32]
  g0[2] = (unsigned int)(gaddr & 0xffffffffu);         // global_addr lo
  g0[3] = (unsigned int)((gaddr >> 32) & 0x01ffffffu)  // global_addr hi (57b)
          | (2u << 30);                                // type=2 ("image")
  v8i_t g1;
  g1[0] = (int)((1u << 16)                 // data_size: 2 bytes
                | (1u << 20)               // pad_enable
                | (pad_int_code << 22)
                | (pad_amt_code << 25));
  g1[1] = (int)(tile_w << 16);             // tensor_dim0 lo16 (bits 63:48)
  g1[2] = (int)((rows_avail & 0xffffu) << 16);  // tensor_dim1 lo16 (bits 95:80)
  g1[3] = (int)(tile_w << 16);             // tile_dim0 (bits 127:112)
  g1[4] = (int)tile_rows;                  // tile_dim1 (bits 143:128)
  g1[5] = (int)stride_elems;               // tensor_dim0_stride lo32
  g1[6] = 0;
  g1[7] = 0;
  v4i_t z4i = {0, 0, 0, 0};
  v8i_t z8i = {0, 0, 0, 0, 0, 0, 0, 0};
  __builtin_amdgcn_tensor_load_to_lds(g0, g1, z4i, z4i, z8i, 0);
}

// =====================================================================
// GEMM computed as C^T tiles: C[M,N] = Act[M,K] @ W[K,N], W pre-transposed
// Wt[N,K].  Double-buffered TDM pipeline (flipL==0): wave 0 issues the
// next tile's DMAs then s_wait_tensorcnt(2) so tile k+1 streams in while
// tile k is consumed by WMMA.  Flipped path uses manual b128 staging.
// =====================================================================
#define GEMM_LDS (4 * 128 * 72 * 2)

__global__ __launch_bounds__(256) void gemm_bf16_kernel(
    const unsigned short* __restrict__ Act, const unsigned short* __restrict__ Wt,
    float* __restrict__ C, const float* __restrict__ bias,
    int M, int N, int K, int flipL) {
  extern __shared__ char smem[];
  const int LDK = 72;
  const int BUFE = 2 * 128 * 72;                // elems per double-buffer slot
  const unsigned int BUFB = (unsigned int)BUFE * 2u;   // bytes per slot
  const unsigned int XOFFB = 128u * 72u * 2u;          // sX byte offset in slot
  unsigned short* lds0 = (unsigned short*)smem;
  int tid = threadIdx.x, lane = tid & 31, wave = tid >> 5;
  int bm = blockIdx.y * 128, bn = blockIdx.x * 128;
  int wn = (wave & 3) * 32, wm = (wave >> 2) * 64;

  v8f z8 = {0.f, 0.f, 0.f, 0.f, 0.f, 0.f, 0.f, 0.f};
  v8f acc[2][4];
#pragma unroll
  for (int i = 0; i < 2; ++i)
#pragma unroll
    for (int j = 0; j < 4; ++j) acc[i][j] = z8;

  // manual staging geometry (used for flipped A rows)
  int rr = tid >> 1, kc = (tid & 1) * 32;
  int r = bm + rr, gr = r;
  if (flipL) {
    int bb = r / flipL;
    int l = r - bb * flipL;
    gr = bb * flipL + (flipL - 1 - l);
  }
  const unsigned short* ArowG = Act + (size_t)gr * K;
  int gn = bn + rr;
  int gnc = gn < N ? gn : N - 1;                 // clamp: keep loads global
  const unsigned short* WrowG = Wt + (size_t)gnc * K;

  unsigned int base_off = (unsigned int)(unsigned long long)(void*)lds0;
  unsigned int rowsW = (unsigned int)(N - bn < 128 ? N - bn : 128);
  unsigned long long wtBase =
      (unsigned long long)(const void*)(Wt) + (((size_t)bn * K) << 1);
  unsigned long long acBase =
      (unsigned long long)(const void*)(Act) + (((size_t)bm * K) << 1);

  if (flipL == 0 && wave == 0) {                 // pipeline prologue: tile 0
    tdm_load_2d(base_off, wtBase, 64u, 128u, rowsW, (unsigned)K, 4u, 3u);
    tdm_load_2d(base_off + XOFFB, acBase, 64u, 128u, 128u, (unsigned)K, 4u, 3u);
  }

  int nbuf = 0;
  for (int kb = 0; kb < K; kb += 64) {
    unsigned short* sW = lds0 + nbuf * BUFE;
    unsigned short* sX = sW + 128 * LDK;
    if (flipL == 0) {
      if (wave == 0) {
        if (kb + 64 < K) {                       // issue next tile, wait current
          unsigned int noff = base_off + (unsigned)(nbuf ^ 1) * BUFB;
          unsigned long long koff = ((size_t)(kb + 64)) << 1;
          tdm_load_2d(noff, wtBase + koff, 64u, 128u, rowsW, (unsigned)K, 4u, 3u);
          tdm_load_2d(noff + XOFFB, acBase + koff, 64u, 128u, 128u, (unsigned)K, 4u, 3u);
          __builtin_amdgcn_s_wait_tensorcnt(2);
        } else {
          __builtin_amdgcn_s_wait_tensorcnt(0);
        }
      }
    } else {
#pragma unroll
      for (int q = 0; q < 4; ++q) {
        *(uint4*)&sX[rr * LDK + kc + q * 8] = *(const uint4*)&ArowG[kb + kc + q * 8];
        *(uint4*)&sW[rr * LDK + kc + q * 8] = *(const uint4*)&WrowG[kb + kc + q * 8];
      }
      if (kb + 64 < K) {
        __builtin_prefetch((const void*)&ArowG[kb + 64 + kc], 0, 0);
        __builtin_prefetch((const void*)&WrowG[kb + 64 + kc], 0, 0);
      }
    }
    __syncthreads();
#pragma unroll
    for (int ks = 0; ks < 64; ks += 32) {
      v16bf a0 = frag_A(sW, wn, ks, LDK);
      v16bf a1 = frag_A(sW, wn + 16, ks, LDK);
#pragma unroll
      for (int j = 0; j < 4; ++j) {
        v16bf bb = frag_Bc(sX, ks, wm + j * 16, LDK);
        acc[0][j] = wmma_bf16(a0, bb, acc[0][j]);
        acc[1][j] = wmma_bf16(a1, bb, acc[1][j]);
      }
    }
    __syncthreads();
    nbuf ^= 1;
  }

  int half = lane >> 4, nn = lane & 15;
#pragma unroll
  for (int i = 0; i < 2; ++i) {
    int n0g = bn + wn + i * 16 + 8 * half;       // 8 consecutive output cols
#pragma unroll
    for (int j = 0; j < 4; ++j) {
      int mg = bm + wm + j * 16 + nn;            // output row
      F4x2 v;
#pragma unroll
      for (int e = 0; e < 8; ++e)
        v.f[e] = acc[i][j][e] +
                 ((bias && n0g + e < N) ? bias[n0g + e] : 0.f);
      float* dst = C + (size_t)mg * N + n0g;
      if (n0g + 7 < N) {
        *(float4*)dst = v.v[0];
        *(float4*)(dst + 4) = v.v[1];
      } else {
#pragma unroll
        for (int e = 0; e < 8; ++e)
          if (n0g + e < N) dst[e] = v.f[e];
      }
    }
  }
}

// =====================================================================
// Elementwise / small kernels
// =====================================================================
__global__ void cvt_kernel(const float* __restrict__ src,
                           unsigned short* __restrict__ dst, int n) {
  int gid = blockIdx.x * 256 + threadIdx.x;
  if (gid < n) dst[gid] = f2bf(src[gid]);
}
// f32 [K,N] -> bf16 transposed [N,K]
__global__ void cvt_t_kernel(const float* __restrict__ src,
                             unsigned short* __restrict__ dst, int K, int N) {
  int gid = blockIdx.x * 256 + threadIdx.x;
  if (gid >= K * N) return;
  int k = gid / N, n = gid - k * N;
  dst[(size_t)n * K + k] = f2bf(src[gid]);
}

// LayerNorm over D_IN, write bf16 h and silu gate
__global__ __launch_bounds__(256) void pre_ln_kernel(
    const float* __restrict__ raw, const float* __restrict__ g,
    const float* __restrict__ be, unsigned short* __restrict__ h_bf,
    float* __restrict__ gate) {
  __shared__ float r1[256], r2[256];
  int row = blockIdx.x, tid = threadIdx.x;
  float v[4], s = 0.f, sq = 0.f;
#pragma unroll
  for (int e = 0; e < 4; ++e) {
    v[e] = raw[(size_t)row * D_IN + tid + e * 256];
    s += v[e];
    sq += v[e] * v[e];
  }
  r1[tid] = s; r2[tid] = sq;
  __syncthreads();
  for (int st = 128; st > 0; st >>= 1) {
    if (tid < st) { r1[tid] += r1[tid + st]; r2[tid] += r2[tid + st]; }
    __syncthreads();
  }
  float mu = r1[0] / (float)D_IN;
  float var = r2[0] / (float)D_IN - mu * mu;
  float rs = rsqrtf(var + 1e-5f);
#pragma unroll
  for (int e = 0; e < 4; ++e) {
    int d = tid + e * 256;
    float hn = (v[e] - mu) * rs * g[d] + be[d];
    h_bf[(size_t)row * D_IN + d] = f2bf(hn);
    gate[(size_t)row * D_IN + d] = hn / (1.f + __expf(-hn));
  }
}

// depthwise causal conv (K=4) + silu; route to xs, xsT, B, C bf16 streams
__global__ void conv_kernel(const float* __restrict__ zx,
                            const float* __restrict__ cw,
                            const float* __restrict__ cb,
                            unsigned short* __restrict__ xs,
                            unsigned short* __restrict__ xsT,
                            unsigned short* __restrict__ Bb,
                            unsigned short* __restrict__ Cb) {
  int gid = blockIdx.x * 256 + threadIdx.x;
  if (gid >= BATCH * SEQ * CONV_DIM) return;
  int ch = gid % CONV_DIM;
  int l = (gid / CONV_DIM) % SEQ;
  int b = gid / (CONV_DIM * SEQ);
  float acc = cb[ch];
#pragma unroll
  for (int k = 0; k < D_CONVK; ++k) {
    int ll = l - (D_CONVK - 1) + k;
    if (ll >= 0)
      acc += cw[ch * D_CONVK + k] *
             zx[((size_t)(b * SEQ + ll)) * D_PROJ + D_INNER + ch];
  }
  float sv = acc / (1.f + __expf(-acc));
  unsigned short bv = f2bf(sv);
  size_t row = (size_t)(b * SEQ + l);
  if (ch < D_INNER) {
    xs[row * D_INNER + ch] = bv;
    int h = ch >> 6, p = ch & 63;
    int c = l >> 7, t = l & 127;
    xsT[((((size_t)(b * NHEADS + h) * NCHUNK + c) * HEADDIM + p) * QCHUNK) + t] = bv;
  } else if (ch < D_INNER + D_STATE) {
    Bb[row * D_STATE + (ch - D_INNER)] = bv;
  } else {
    Cb[row * D_STATE + (ch - D_INNER - D_STATE)] = bv;
  }
}

// dt = softplus(raw + dtb) ; dtA = dt * (-exp(Alog))
__global__ void dt_kernel(const float* __restrict__ zx,
                          const float* __restrict__ dtb,
                          const float* __restrict__ Alog,
                          float* __restrict__ sdt_o, float* __restrict__ dtA_o) {
  int gid = blockIdx.x * 256 + threadIdx.x;
  if (gid >= NTOK * NHEADS) return;
  int h = gid % NHEADS;
  size_t row = gid / NHEADS;
  float raw = zx[row * D_PROJ + D_INNER + CONV_DIM + h] + dtb[h];
  float dt = (raw > 20.f) ? raw : log1pf(__expf(raw));
  sdt_o[gid] = dt;
  dtA_o[gid] = -__expf(Alog[h]) * dt;
}

// =====================================================================
// SSD intra-chunk, per (chunk, head, batch), 8 waves.  Tile staging via
// TDM (B/C: 128x128, X^T: 64x128; LDS rows padded to 136 elems).
// =====================================================================
#define SSD_INTRA_LDS (256 * 4 + (128 * 136 * 2 + 64 * 136) * 2)

__global__ __launch_bounds__(256) void ssd_intra_kernel(
    const unsigned short* __restrict__ xsT, const unsigned short* __restrict__ Bb,
    const unsigned short* __restrict__ Cb, const float* __restrict__ dtAg,
    const float* __restrict__ sdtg, float* __restrict__ y,
    unsigned short* __restrict__ states, float* __restrict__ cumd,
    float* __restrict__ ctot) {
  extern __shared__ char smem[];
  float* scum = (float*)smem;                             // 128
  float* sdtl = scum + 128;                               // 128
  unsigned short* sBm = (unsigned short*)(sdtl + 128);    // 128 t x 136
  unsigned short* sCm = sBm + 128 * 136;                  // 128 t x 136
  unsigned short* sXt = sCm + 128 * 136;                  // 64 p x 136
  const int LD = 136;

  int tid = threadIdx.x, lane = tid & 31, wave = tid >> 5;
  int c = blockIdx.x, h = blockIdx.y, b = blockIdx.z;
  int l0 = c * QCHUNK;
  size_t rowbase = (size_t)(b * SEQ + l0);
  size_t xtbase = (((size_t)(b * NHEADS + h) * NCHUNK + c) * HEADDIM) * QCHUNK;

  if (wave == 0) {       // one wave drives all three tile DMAs
    tdm_load_2d((unsigned int)(unsigned long long)(void*)sBm,
                (unsigned long long)(const void*)(Bb + rowbase * D_STATE),
                128u, 128u, 128u, 128u, 5u, 3u);
    tdm_load_2d((unsigned int)(unsigned long long)(void*)sCm,
                (unsigned long long)(const void*)(Cb + rowbase * D_STATE),
                128u, 128u, 128u, 128u, 5u, 3u);
    tdm_load_2d((unsigned int)(unsigned long long)(void*)sXt,
                (unsigned long long)(const void*)(xsT + xtbase),
                128u, 64u, 64u, 128u, 5u, 3u);
    __builtin_amdgcn_s_wait_tensorcnt(0);
  }
  if (tid < 128) {
    scum[tid] = dtAg[(rowbase + tid) * NHEADS + h];
    sdtl[tid] = sdtg[(rowbase + tid) * NHEADS + h];
  }
  __syncthreads();
  for (int offd = 1; offd < 128; offd <<= 1) {
    float v = 0.f;
    if (tid < 128 && tid >= offd) v = scum[tid - offd];
    __syncthreads();
    if (tid < 128 && tid >= offd) scum[tid] += v;
    __syncthreads();
  }
  if (tid < 128) cumd[(rowbase + tid) * NHEADS + h] = scum[tid];
  if (tid == 127) ctot[(b * NHEADS + h) * NCHUNK + c] = __expf(scum[127]);
  __syncthreads();

  v8f z8 = {0.f, 0.f, 0.f, 0.f, 0.f, 0.f, 0.f, 0.f};
  int half = lane >> 4, nn = lane & 15;

  // --- Stage A: G^T[s][t] ---
  int s0 = wave * 16;
  v8f g[8];
#pragma unroll
  for (int j = 0; j < 8; ++j) g[j] = z8;
#pragma unroll
  for (int kk = 0; kk < 4; ++kk) {
    v16bf a = frag_A(sBm, s0, kk * 32, LD);
#pragma unroll
    for (int j = 0; j < 8; ++j) {
      v16bf bb = frag_Bc(sCm, kk * 32, j * 16, LD);
      g[j] = wmma_bf16(a, bb, g[j]);
    }
  }
  __syncthreads();

  // write M[t][s] (uint4), scale B in place
#pragma unroll
  for (int j = 0; j < 8; ++j) {
    int t = j * 16 + nn;
    PK8 pk;
#pragma unroll
    for (int e = 0; e < 8; ++e) {
      int s = s0 + 8 * half + e;
      float m = 0.f;
      if (s <= t) m = g[j][e] * __expf(scum[t] - scum[s]) * sdtl[s];
      pk.us[e] = f2bf(m);
    }
    *(uint4*)&sCm[t * LD + s0 + 8 * half] = pk.q;
  }
  {
    float send = scum[127];
    for (int i = tid; i < 128 * 64; i += 256) {
      int t = i >> 6, n2 = i & 63;
      float sc = __expf(send - scum[t]) * sdtl[t];
      unsigned int v = *(unsigned int*)&sBm[t * LD + n2 * 2];
      unsigned int o = (unsigned int)f2bf(bf2f((unsigned short)(v & 0xffffu)) * sc) |
                       ((unsigned int)f2bf(bf2f((unsigned short)(v >> 16)) * sc) << 16);
      *(unsigned int*)&sBm[t * LD + n2 * 2] = o;
    }
  }
  __syncthreads();

  int p0 = (wave & 3) * 16, chalf = (wave >> 2) * 64;

  // --- Stage B: Y^T[p][t] = X^T @ M^T ---
  {
    v8f ya[4];
#pragma unroll
    for (int j = 0; j < 4; ++j) ya[j] = z8;
#pragma unroll
    for (int kk = 0; kk < 4; ++kk) {
      v16bf a = frag_A(sXt, p0, kk * 32, LD);
#pragma unroll
      for (int j = 0; j < 4; ++j) {
        v16bf bb = frag_Bc(sCm, kk * 32, chalf + j * 16, LD);
        ya[j] = wmma_bf16(a, bb, ya[j]);
      }
    }
#pragma unroll
    for (int j = 0; j < 4; ++j) {
      int t = chalf + j * 16 + nn;
      F4x2 v;
#pragma unroll
      for (int e = 0; e < 8; ++e) v.f[e] = ya[j][e];
      float* dst = y + (rowbase + t) * D_INNER + h * HEADDIM + p0 + 8 * half;
      *(float4*)dst = v.v[0];
      *(float4*)(dst + 4) = v.v[1];
    }
  }

  // --- Stage C: S^T tiles -> states[n][p] ---
  {
    v8f sa[4];
#pragma unroll
    for (int j = 0; j < 4; ++j) sa[j] = z8;
#pragma unroll
    for (int kk = 0; kk < 4; ++kk) {
      v16bf a = frag_A(sXt, p0, kk * 32, LD);
#pragma unroll
      for (int j = 0; j < 4; ++j) {
        BF16x16 bf;
#pragma unroll
        for (int e = 0; e < 16; ++e)
          bf.u[e] = sBm[(kk * 32 + (e < 8 ? 8 * half + e : 16 + 8 * half + e - 8)) * LD + chalf + j * 16 + nn];
        v16bf bb = bf.v;
        sa[j] = wmma_bf16(a, bb, sa[j]);
      }
    }
    size_t sb = ((size_t)(b * NHEADS + h) * NCHUNK + c) * (D_STATE * HEADDIM);
#pragma unroll
    for (int j = 0; j < 4; ++j) {
      int n = chalf + j * 16 + nn;
      PK8 pk;
#pragma unroll
      for (int e = 0; e < 8; ++e) pk.us[e] = f2bf(sa[j][e]);
      *(uint4*)&states[sb + n * HEADDIM + p0 + 8 * half] = pk.q;
    }
  }
}

// sequential inter-chunk recurrence; transposes [n][p] -> [p][n] in place
__global__ __launch_bounds__(256) void state_scan_kernel(
    unsigned short* __restrict__ states, const float* __restrict__ ctot) {
  int bh = blockIdx.x, tid = threadIdx.x;
  size_t base = (size_t)bh * NCHUNK * D_STATE * HEADDIM;
  float H[32];
#pragma unroll
  for (int e = 0; e < 32; ++e) H[e] = 0.f;
  for (int c = 0; c < NCHUNK; ++c) {
    float ct = ctot[bh * NCHUNK + c];
    size_t cb = base + (size_t)c * D_STATE * HEADDIM;
    float s[32];
#pragma unroll
    for (int e = 0; e < 32; ++e) s[e] = bf2f(states[cb + tid + e * 256]);
    __syncthreads();
#pragma unroll
    for (int e = 0; e < 32; ++e) {
      int idx = tid + e * 256, n = idx >> 6, p = idx & 63;
      states[cb + p * D_STATE + n] = f2bf(H[e]);
      H[e] = H[e] * ct + s[e];
    }
    __syncthreads();
  }
}

// Y^T += H^T @ Cs^T  (+ D*x); H^T tile staged via TDM
#define SSD_INTER_LDS (128 * 4 + (128 * 136 + 64 * 136) * 2)

__global__ __launch_bounds__(256) void ssd_inter_kernel(
    const unsigned short* __restrict__ Cb, const unsigned short* __restrict__ states,
    const float* __restrict__ cumd, const unsigned short* __restrict__ xs,
    const float* __restrict__ Dp, float* __restrict__ y) {
  extern __shared__ char smem[];
  float* ed = (float*)smem;                            // 128
  unsigned short* sCs = (unsigned short*)(ed + 128);   // 128 t x 136
  unsigned short* sHt = sCs + 128 * 136;               // 64 p x 136
  const int LD = 136;
  int tid = threadIdx.x, lane = tid & 31, wave = tid >> 5;
  int c = blockIdx.x, h = blockIdx.y, b = blockIdx.z;
  int l0 = c * QCHUNK;
  size_t rowbase = (size_t)(b * SEQ + l0);
  size_t sb = ((size_t)(b * NHEADS + h) * NCHUNK + c) * (D_STATE * HEADDIM);

  if (wave == 0) {                                     // H^T [p][n] via TDM
    tdm_load_2d((unsigned int)(unsigned long long)(void*)sHt,
                (unsigned long long)(const void*)(states + sb),
                128u, 64u, 64u, 128u, 5u, 3u);
    __builtin_amdgcn_s_wait_tensorcnt(0);
  }
  if (tid < 128) ed[tid] = __expf(cumd[(rowbase + tid) * NHEADS + h]);
  __syncthreads();
  const unsigned int* C32 = (const unsigned int*)Cb;
  for (int i = tid; i < 128 * 64; i += 256) {          // Cs[t][n] scaled
    int t = i >> 6, n2 = i & 63;
    unsigned int v = C32[(rowbase + t) * (D_STATE / 2) + n2];
    float sc = ed[t];
    unsigned int o = (unsigned int)f2bf(bf2f((unsigned short)(v & 0xffffu)) * sc) |
                     ((unsigned int)f2bf(bf2f((unsigned short)(v >> 16)) * sc) << 16);
    *(unsigned int*)&sCs[t * LD + n2 * 2] = o;
  }
  __syncthreads();

  int half = lane >> 4, nn = lane & 15;
  int p0 = (wave & 3) * 16, thalf = (wave >> 2) * 64;
  v8f z8 = {0.f, 0.f, 0.f, 0.f, 0.f, 0.f, 0.f, 0.f};
  v8f acc[4];
#pragma unroll
  for (int j = 0; j < 4; ++j) acc[j] = z8;
#pragma unroll
  for (int kk = 0; kk < 4; ++kk) {
    v16bf a = frag_A(sHt, p0, kk * 32, LD);
#pragma unroll
    for (int j = 0; j < 4; ++j) {
      v16bf bb = frag_Bc(sCs, kk * 32, thalf + j * 16, LD);
      acc[j] = wmma_bf16(a, bb, acc[j]);
    }
  }
  float Dh = Dp[h];
#pragma unroll
  for (int j = 0; j < 4; ++j) {
    int t = thalf + j * 16 + nn;
    float* dst = y + (rowbase + t) * D_INNER + h * HEADDIM + p0 + 8 * half;
    const unsigned short* xsp =
        xs + (rowbase + t) * D_INNER + h * HEADDIM + p0 + 8 * half;
    F4x2 v;
    v.v[0] = *(const float4*)dst;
    v.v[1] = *(const float4*)(dst + 4);
    PK8 xv;
    xv.q = *(const uint4*)xsp;
#pragma unroll
    for (int e = 0; e < 8; ++e) v.f[e] += acc[j][e] + Dh * bf2f(xv.us[e]);
    *(float4*)dst = v.v[0];
    *(float4*)(dst + 4) = v.v[1];
  }
}

// gated RMS norm over D_INNER -> bf16
__global__ __launch_bounds__(256) void gate_rms_kernel(
    const float* __restrict__ y, const float* __restrict__ zx,
    const float* __restrict__ nw, unsigned short* __restrict__ yn) {
  __shared__ float r1[256];
  int row = blockIdx.x, tid = threadIdx.x;
  float v[8], sq = 0.f;
#pragma unroll
  for (int e = 0; e < 8; ++e) {
    int d = tid + e * 256;
    float z = zx[(size_t)row * D_PROJ + d];
    float g = y[(size_t)row * D_INNER + d] * (z / (1.f + __expf(-z)));
    v[e] = g;
    sq += g * g;
  }
  r1[tid] = sq;
  __syncthreads();
  for (int st = 128; st > 0; st >>= 1) {
    if (tid < st) r1[tid] += r1[tid + st];
    __syncthreads();
  }
  float rs = rsqrtf(r1[0] / (float)D_INNER + 1e-5f);
#pragma unroll
  for (int e = 0; e < 8; ++e) {
    int d = tid + e * 256;
    yn[(size_t)row * D_INNER + d] = f2bf(v[e] * rs * nw[d]);
  }
}

// comb = (out_f + flip(out_b)) * gate -> bf16
__global__ void combine_kernel(const float* __restrict__ of,
                               const float* __restrict__ ob,
                               const float* __restrict__ gate,
                               unsigned short* __restrict__ comb) {
  int gid = blockIdx.x * 256 + threadIdx.x;
  if (gid >= NTOK * D_IN) return;
  int d = gid % D_IN;
  int l = (gid / D_IN) % SEQ;
  int b = gid / (D_IN * SEQ);
  float v = of[gid] + ob[((size_t)(b * SEQ + (SEQ - 1 - l))) * D_IN + d];
  comb[gid] = f2bf(v * gate[gid]);
}

// final LayerNorm over D_MODEL + residual
__global__ __launch_bounds__(256) void post_ln_kernel(
    const float* __restrict__ raw, const float* __restrict__ g,
    const float* __restrict__ be, const float* __restrict__ x,
    float* __restrict__ out) {
  __shared__ float r1[256], r2[256];
  int row = blockIdx.x, tid = threadIdx.x;
  float v[2], s = 0.f, sq = 0.f;
#pragma unroll
  for (int e = 0; e < 2; ++e) {
    v[e] = raw[(size_t)row * D_MODEL + tid + e * 256];
    s += v[e];
    sq += v[e] * v[e];
  }
  r1[tid] = s; r2[tid] = sq;
  __syncthreads();
  for (int st = 128; st > 0; st >>= 1) {
    if (tid < st) { r1[tid] += r1[tid + st]; r2[tid] += r2[tid + st]; }
    __syncthreads();
  }
  float mu = r1[0] / (float)D_MODEL;
  float var = r2[0] / (float)D_MODEL - mu * mu;
  float rs = rsqrtf(var + 1e-5f);
#pragma unroll
  for (int e = 0; e < 2; ++e) {
    int d = tid + e * 256;
    out[(size_t)row * D_MODEL + d] =
        (v[e] - mu) * rs * g[d] + be[d] + x[(size_t)row * D_MODEL + d];
  }
}

// =====================================================================
// Host orchestration
// =====================================================================
extern "C" void kernel_launch(void* const* d_in, const int* in_sizes, int n_in,
                              void* d_out, int out_size, void* d_ws, size_t ws_size,
                              hipStream_t stream) {
  (void)in_sizes; (void)n_in; (void)out_size; (void)ws_size;
  const float* x        = (const float*)d_in[0];
  const float* pre_w    = (const float*)d_in[1];
  const float* pre_b    = (const float*)d_in[2];
  const float* pre_ln_g = (const float*)d_in[3];
  const float* pre_ln_b = (const float*)d_in[4];
  const float* post_w   = (const float*)d_in[5];
  const float* post_b   = (const float*)d_in[6];
  const float* post_ln_g= (const float*)d_in[7];
  const float* post_ln_b= (const float*)d_in[8];
  const float* win_d[2]   = {(const float*)d_in[9],  (const float*)d_in[17]};
  const float* convw_d[2] = {(const float*)d_in[10], (const float*)d_in[18]};
  const float* convb_d[2] = {(const float*)d_in[11], (const float*)d_in[19]};
  const float* dtb_d[2]   = {(const float*)d_in[12], (const float*)d_in[20]};
  const float* Alog_d[2]  = {(const float*)d_in[13], (const float*)d_in[21]};
  const float* Dp_d[2]    = {(const float*)d_in[14], (const float*)d_in[22]};
  const float* normw_d[2] = {(const float*)d_in[15], (const float*)d_in[23]};
  const float* wout_d[2]  = {(const float*)d_in[16], (const float*)d_in[24]};
  float* out = (float*)d_out;

  char* ws = (char*)d_ws;
  size_t off = 0;
  auto alloc = [&](size_t bytes) {
    size_t r = off;
    off += (bytes + 255) & ~(size_t)255;
    return r;
  };
  unsigned short* x_bf  = (unsigned short*)(ws + alloc((size_t)NTOK * D_MODEL * 2));
  unsigned short* h_bf  = (unsigned short*)(ws + alloc((size_t)NTOK * D_IN * 2));
  float* gate           = (float*)(ws + alloc((size_t)NTOK * D_IN * 4));
  unsigned short* wpre  = (unsigned short*)(ws + alloc((size_t)D_MODEL * D_IN * 2));
  unsigned short* wwin0 = (unsigned short*)(ws + alloc((size_t)D_IN * D_PROJ * 2));
  unsigned short* wwin1 = (unsigned short*)(ws + alloc((size_t)D_IN * D_PROJ * 2));
  unsigned short* wwo0  = (unsigned short*)(ws + alloc((size_t)D_INNER * D_IN * 2));
  unsigned short* wwo1  = (unsigned short*)(ws + alloc((size_t)D_INNER * D_IN * 2));
  unsigned short* wpost = (unsigned short*)(ws + alloc((size_t)D_IN * D_MODEL * 2));
  float* zxbcdt         = (float*)(ws + alloc((size_t)NTOK * D_PROJ * 4));
  unsigned short* xs    = (unsigned short*)(ws + alloc((size_t)NTOK * D_INNER * 2));
  unsigned short* xsT   = (unsigned short*)(ws + alloc((size_t)NTOK * D_INNER * 2));
  unsigned short* Bb    = (unsigned short*)(ws + alloc((size_t)NTOK * D_STATE * 2));
  unsigned short* Cb    = (unsigned short*)(ws + alloc((size_t)NTOK * D_STATE * 2));
  float* sdt_ws         = (float*)(ws + alloc((size_t)NTOK * NHEADS * 4));
  float* dtA_ws         = (float*)(ws + alloc((size_t)NTOK * NHEADS * 4));
  float* cumd           = (float*)(ws + alloc((size_t)NTOK * NHEADS * 4));
  float* ctot           = (float*)(ws + alloc((size_t)BATCH * NHEADS * NCHUNK * 4));
  unsigned short* states= (unsigned short*)(ws + alloc((size_t)BATCH * NHEADS * NCHUNK * D_STATE * HEADDIM * 2));
  float* ybuf           = (float*)(ws + alloc((size_t)NTOK * D_INNER * 4));
  unsigned short* yn    = (unsigned short*)(ws + alloc((size_t)NTOK * D_INNER * 2));
  float* outf           = (float*)(ws + alloc((size_t)NTOK * D_IN * 4));
  float* outb           = (float*)(ws + alloc((size_t)NTOK * D_IN * 4));
  float* pre_raw        = ybuf;
  unsigned short* comb  = h_bf;
  float* post_raw       = zxbcdt;
  unsigned short* wwin_a[2] = {wwin0, wwin1};
  unsigned short* wwo_a[2]  = {wwo0, wwo1};
  float* out_dir[2] = {outf, outb};

  auto cvt = [&](const float* s, unsigned short* d, int n) {
    cvt_kernel<<<(n + 255) / 256, 256, 0, stream>>>(s, d, n);
  };
  auto cvt_t = [&](const float* s, unsigned short* d, int K, int N) {
    cvt_t_kernel<<<(K * N + 255) / 256, 256, 0, stream>>>(s, d, K, N);
  };
  cvt(x, x_bf, NTOK * D_MODEL);
  cvt_t(pre_w, wpre, D_MODEL, D_IN);
  cvt_t(win_d[0], wwin0, D_IN, D_PROJ);
  cvt_t(win_d[1], wwin1, D_IN, D_PROJ);
  cvt_t(wout_d[0], wwo0, D_INNER, D_IN);
  cvt_t(wout_d[1], wwo1, D_INNER, D_IN);
  cvt_t(post_w, wpost, D_IN, D_MODEL);

  gemm_bf16_kernel<<<dim3(D_IN / 128, NTOK / 128), 256, GEMM_LDS, stream>>>(
      x_bf, wpre, pre_raw, pre_b, NTOK, D_IN, D_MODEL, 0);
  pre_ln_kernel<<<NTOK, 256, 0, stream>>>(pre_raw, pre_ln_g, pre_ln_b, h_bf, gate);

  for (int d = 0; d < 2; ++d) {
    gemm_bf16_kernel<<<dim3((D_PROJ + 127) / 128, NTOK / 128), 256, GEMM_LDS, stream>>>(
        h_bf, wwin_a[d], zxbcdt, nullptr, NTOK, D_PROJ, D_IN, d ? SEQ : 0);
    conv_kernel<<<(NTOK * CONV_DIM + 255) / 256, 256, 0, stream>>>(
        zxbcdt, convw_d[d], convb_d[d], xs, xsT, Bb, Cb);
    dt_kernel<<<(NTOK * NHEADS + 255) / 256, 256, 0, stream>>>(
        zxbcdt, dtb_d[d], Alog_d[d], sdt_ws, dtA_ws);
    ssd_intra_kernel<<<dim3(NCHUNK, NHEADS, BATCH), 256, SSD_INTRA_LDS, stream>>>(
        xsT, Bb, Cb, dtA_ws, sdt_ws, ybuf, states, cumd, ctot);
    state_scan_kernel<<<BATCH * NHEADS, 256, 0, stream>>>(states, ctot);
    ssd_inter_kernel<<<dim3(NCHUNK, NHEADS, BATCH), 256, SSD_INTER_LDS, stream>>>(
        Cb, states, cumd, xs, Dp_d[d], ybuf);
    gate_rms_kernel<<<NTOK, 256, 0, stream>>>(ybuf, zxbcdt, normw_d[d], yn);
    gemm_bf16_kernel<<<dim3(D_IN / 128, NTOK / 128), 256, GEMM_LDS, stream>>>(
        yn, wwo_a[d], out_dir[d], nullptr, NTOK, D_IN, D_INNER, 0);
  }

  combine_kernel<<<(NTOK * D_IN + 255) / 256, 256, 0, stream>>>(outf, outb, gate, comb);
  gemm_bf16_kernel<<<dim3(D_MODEL / 128, NTOK / 128), 256, GEMM_LDS, stream>>>(
      comb, wpost, post_raw, post_b, NTOK, D_MODEL, D_IN, 0);
  post_ln_kernel<<<NTOK, 256, 0, stream>>>(post_raw, post_ln_g, post_ln_b, x, out);
}
